// Encoder_34170759807168
// MI455X (gfx1250) — compile-verified, matrix operand order checked by reading, not run on hardware
//
#include <hip/hip_runtime.h>

#define S_LEN 64
#define BATCH 128
#define WLEN  16
#define E_W   300
#define E_C   64
#define H_C   128
#define H_W   512
#define KPAD_WIH 576   // 556 padded to multiple of 32

typedef _Float16 v16h __attribute__((ext_vector_type(16)));
typedef _Float16 f16x8 __attribute__((ext_vector_type(8)));
typedef float    v8f  __attribute__((ext_vector_type(8)));

__device__ __forceinline__ float sigm(float x) { return 1.f / (1.f + __expf(-x)); }
__device__ __forceinline__ float tanh_(float x) {
  float e = __expf(-2.f * x);
  return (1.f - e) / (1.f + e);
}

__device__ __forceinline__ v8f wmma_f16(v16h a, v16h b, v8f c) {
  return __builtin_amdgcn_wmma_f32_16x16x32_f16(false, a, false, b, (short)0, c, false, false);
}

// ---- gfx1250 async Global->LDS DMA (ASYNCcnt-tracked, bypasses VGPRs) ------
__device__ __forceinline__ void async_g2l_b128(unsigned lds_addr, const void* gaddr) {
  asm volatile("global_load_async_to_lds_b128 %0, %1, off"
               :: "v"(lds_addr), "v"(gaddr) : "memory");
}
__device__ __forceinline__ void async_g2l_b64(unsigned lds_addr, const void* gaddr) {
  asm volatile("global_load_async_to_lds_b64 %0, %1, off"
               :: "v"(lds_addr), "v"(gaddr) : "memory");
}
__device__ __forceinline__ void wait_async0() {
  asm volatile("s_wait_asynccnt 0x0" ::: "memory");
}
// Generic LDS pointer -> wave LDS byte offset (flat aperture: LDS = addr[31:0])
__device__ __forceinline__ unsigned lds_off(const void* p) {
  return (unsigned)(uintptr_t)p;
}

// A fragment (16x32 f16, M x K) from row-major [16][kstride] f16 buffer.
// ISA layout: lane<16 holds row=lane, K k0..k0+7 (halves 0-7) and k0+16..k0+23
// (halves 8-15); lane>=16 holds row=lane-16, K k0+8..15 and k0+24..31.
// => two contiguous 16B loads at off and off+16.
__device__ __forceinline__ v16h load_a_frag(const _Float16* base, int kstride, int k0) {
  int lane = threadIdx.x & 31;
  int off = (lane & 15) * kstride + k0 + ((lane >= 16) ? 8 : 0);
  union { v16h v; f16x8 h[2]; } u;
  u.h[0] = *(const f16x8*)(base + off);
  u.h[1] = *(const f16x8*)(base + off + 16);
  return u.v;
}

// B fragment from pre-packed weights: one 32-byte contiguous load per lane.
__device__ __forceinline__ v16h load_b_frag(const _Float16* pk, int tile_idx) {
  int lane = threadIdx.x & 31;
  return *(const v16h*)(pk + (size_t)tile_idx * 512 + lane * 16);
}

// ---------------------------------------------------------------------------
// Pack fp32 weight W[N][K] (row-major) into per-fragment f16 layout:
// dst[((nt*(Kpad/32)+ks)*32 + lane)*16 + half], zero-padding K -> Kpad.
// B 32x16 ISA striping: lane<16 -> col n0+lane, halves 0-7 = K k0..k0+7,
// halves 8-15 = K k0+16..23; lane>=16 -> +8 on the K-within offset.
// ---------------------------------------------------------------------------
__global__ __launch_bounds__(256) void pack_b_kernel(
    const float* __restrict__ src, _Float16* __restrict__ dst,
    int N, int K, int Kpad) {
  int ksteps = Kpad >> 5;
  int total = (N >> 4) * ksteps * 512;
  for (int i = blockIdx.x * blockDim.x + threadIdx.x; i < total;
       i += gridDim.x * blockDim.x) {
    int h = i & 15;
    int lane = (i >> 4) & 31;
    int ts = i >> 9;
    int ks = ts % ksteps;
    int nt = ts / ksteps;
    int n = nt * 16 + (lane & 15);
    int kw = (h < 8 ? h : h + 8) + ((lane >= 16) ? 8 : 0);
    int k = ks * 32 + kw;
    dst[i] = (k < K) ? (_Float16)src[(size_t)n * K + k] : (_Float16)0.f;
  }
}

__global__ __launch_bounds__(256) void conv_f16_kernel(
    const float* __restrict__ s, _Float16* __restrict__ d, int n) {
  for (int i = blockIdx.x * blockDim.x + threadIdx.x; i < n;
       i += gridDim.x * blockDim.x)
    d[i] = (_Float16)s[i];
}

// ---------------------------------------------------------------------------
// Char bi-LSTM: grid (512 row-tiles, 2 dirs), 256 threads (8 waves).
// Both gate weight matrices LDS-resident (192 KB of the 320 KB WGP LDS),
// staged with async Global->LDS DMA. Wave w owns j-tile w (16 cols of the
// 128-wide hidden); per step it computes its i/f/g/o 16x16 tiles:
// 2 WMMAs vs x (K=64) + 4 WMMAs vs h (K=128) each. Only the final hidden
// state is needed (matches reference hT usage).
// ---------------------------------------------------------------------------
__global__ __launch_bounds__(256) void char_lstm_kernel(
    const int* __restrict__ chars, const _Float16* __restrict__ cemb16,
    const _Float16* __restrict__ pkCihF, const _Float16* __restrict__ pkChhF,
    const _Float16* __restrict__ pkCihB, const _Float16* __restrict__ pkChhB,
    const float* __restrict__ cbF, const float* __restrict__ cbB,
    _Float16* __restrict__ cfeat) {
  extern __shared__ _Float16 smem[];
  _Float16* wih = smem;                  // 32768 halves (512x64)
  _Float16* whh = smem + 32768;          // 65536 halves (512x128)
  _Float16* hl  = smem + 32768 + 65536;  // 16x128
  _Float16* xl  = hl + 16 * H_C;         // 16x64

  const int dir = blockIdx.y;
  const _Float16* gih = dir ? pkCihB : pkCihF;
  const _Float16* ghh = dir ? pkChhB : pkChhF;
  const float* cb = dir ? cbB : cbF;

  // Async DMA weight stage: 192 KB straight into LDS (no VGPR round-trip).
  {
    const unsigned wih_l = lds_off(wih);
    const unsigned whh_l = lds_off(whh);
    for (int i = threadIdx.x; i < 4096; i += 256)       // 4096 x 16B
      async_g2l_b128(wih_l + i * 16, (const char*)gih + (size_t)i * 16);
    for (int i = threadIdx.x; i < 8192; i += 256)       // 8192 x 16B
      async_g2l_b128(whh_l + i * 16, (const char*)ghh + (size_t)i * 16);
  }
  for (int i = threadIdx.x; i < 16 * H_C; i += 256) hl[i] = (_Float16)0.f;
  wait_async0();
  __syncthreads();

  const int wave = threadIdx.x >> 5;
  const int lane = threadIdx.x & 31;
  const int col = lane & 15;
  const int rowoff = (lane >> 4) << 3;  // C/D layout: VGPR r -> row r (+8 hi lanes)
  const int n0 = blockIdx.x * 16;

  const float bi = cb[0 * H_C + wave * 16 + col];
  const float bf = cb[1 * H_C + wave * 16 + col];
  const float bg = cb[2 * H_C + wave * 16 + col];
  const float bo = cb[3 * H_C + wave * 16 + col];

  v8f cstate;
#pragma unroll
  for (int r = 0; r < 8; r++) cstate[r] = 0.f;
  float hlast[8];

  const int gx = threadIdx.x * 4;  // each thread DMAs 4 halves (8B) of x
  const int grow = gx >> 6;
  const int gcol = gx & 63;
  const unsigned xl_l = lds_off(xl);

  for (int t = 0; t < WLEN; ++t) {
    const int pos = dir ? (WLEN - 1 - t) : t;
    {
      int cid = chars[(n0 + grow) * WLEN + pos];
      async_g2l_b64(xl_l + (grow * 64 + gcol) * 2, cemb16 + cid * 64 + gcol);
    }
    wait_async0();
    __syncthreads();

    v8f ai, af2, ag, ao;
#pragma unroll
    for (int r = 0; r < 8; r++) { ai[r] = bi; af2[r] = bf; ag[r] = bg; ao[r] = bo; }

#pragma unroll
    for (int k = 0; k < 2; k++) {
      v16h a = load_a_frag(xl, 64, k * 32);
      ai  = wmma_f16(a, load_b_frag(wih, (0 * 8 + wave) * 2 + k), ai);
      af2 = wmma_f16(a, load_b_frag(wih, (1 * 8 + wave) * 2 + k), af2);
      ag  = wmma_f16(a, load_b_frag(wih, (2 * 8 + wave) * 2 + k), ag);
      ao  = wmma_f16(a, load_b_frag(wih, (3 * 8 + wave) * 2 + k), ao);
    }
#pragma unroll
    for (int k = 0; k < 4; k++) {
      v16h a = load_a_frag(hl, H_C, k * 32);
      ai  = wmma_f16(a, load_b_frag(whh, (0 * 8 + wave) * 4 + k), ai);
      af2 = wmma_f16(a, load_b_frag(whh, (1 * 8 + wave) * 4 + k), af2);
      ag  = wmma_f16(a, load_b_frag(whh, (2 * 8 + wave) * 4 + k), ag);
      ao  = wmma_f16(a, load_b_frag(whh, (3 * 8 + wave) * 4 + k), ao);
    }

#pragma unroll
    for (int r = 0; r < 8; r++) {
      float ci = cstate[r];
      ci = sigm(af2[r]) * ci + sigm(ai[r]) * tanh_(ag[r]);
      cstate[r] = ci;
      hlast[r] = sigm(ao[r]) * tanh_(ci);
    }
    __syncthreads();  // all K-reads of h done before overwrite
#pragma unroll
    for (int r = 0; r < 8; r++)
      hl[(r + rowoff) * H_C + wave * 16 + col] = (_Float16)hlast[r];
    __syncthreads();
  }

#pragma unroll
  for (int r = 0; r < 8; r++)
    cfeat[(size_t)(n0 + r + rowoff) * 256 + dir * H_C + wave * 16 + col] =
        (_Float16)hlast[r];
}

// ---------------------------------------------------------------------------
// Word input GEMM: xg = [word_emb | char_feat] @ wW_ih^T + wb  (once, shared by
// both directions). Grid (512 M-tiles, 16 N-chunks), wave w -> n-tile ny*8+w.
// A (16 x 576 f16) staged in LDS, B fragments streamed from L2-resident pack.
// ---------------------------------------------------------------------------
__global__ __launch_bounds__(256) void word_xg_kernel(
    const int* __restrict__ words, const float* __restrict__ wemb,
    const _Float16* __restrict__ cfeat, const _Float16* __restrict__ pkWih,
    const float* __restrict__ wb, float* __restrict__ xg) {
  __shared__ _Float16 al[16 * KPAD_WIH];
  const int mt = blockIdx.x;
  const int ny = blockIdx.y;

  for (int idx = threadIdx.x; idx < 16 * KPAD_WIH; idx += 256) {
    int row = idx / KPAD_WIH, cc = idx % KPAD_WIH;
    int m = mt * 16 + row;           // m = s*B + b
    int s = m >> 7, b = m & 127;
    _Float16 v;
    if (cc < E_W)            v = (_Float16)wemb[(size_t)words[m] * E_W + cc];
    else if (cc < E_W + 256) v = cfeat[(size_t)(b * S_LEN + s) * 256 + (cc - E_W)];
    else                     v = (_Float16)0.f;
    al[row * KPAD_WIH + cc] = v;
  }
  __syncthreads();

  const int wave = threadIdx.x >> 5;
  const int lane = threadIdx.x & 31;
  const int col = lane & 15;
  const int rowoff = (lane >> 4) << 3;
  const int nt = ny * 8 + wave;  // 0..127 over the 2048 gate cols

  const float bv = wb[nt * 16 + col];
  v8f acc;
#pragma unroll
  for (int r = 0; r < 8; r++) acc[r] = bv;

#pragma unroll 6
  for (int k = 0; k < KPAD_WIH / 32; k++) {
    v16h a = load_a_frag(al, KPAD_WIH, k * 32);
    acc = wmma_f16(a, load_b_frag(pkWih, nt * (KPAD_WIH / 32) + k), acc);
  }

  const int m0 = mt * 16;
#pragma unroll
  for (int r = 0; r < 8; r++)
    xg[(size_t)(m0 + r + rowoff) * 2048 + nt * 16 + col] = acc[r];
}

// ---------------------------------------------------------------------------
// Word recurrence: grid (8 batch-tiles, 2 dirs) persistent workgroups; batch
// rows are independent so no cross-WG sync is ever needed. Per step each wave
// handles 4 j-tiles x 4 gates x 16 K-steps = 256 WMMAs against L2-resident
// packed Whh (2 MB, hot in the 192 MB L2). h round-trips through 16 KB LDS.
// Backward writes at mirrored s, matching hs_b[::-1] / cs_b[::-1].
// ---------------------------------------------------------------------------
__global__ __launch_bounds__(256) void word_lstm_kernel(
    const float* __restrict__ xg, const _Float16* __restrict__ pkWhh,
    float* __restrict__ out_h, float* __restrict__ out_c) {
  __shared__ _Float16 hl[16 * H_W];
  const int bt = blockIdx.x;   // batch tile
  const int dir = blockIdx.y;
  for (int i = threadIdx.x; i < 16 * H_W; i += 256) hl[i] = (_Float16)0.f;
  __syncthreads();

  const int wave = threadIdx.x >> 5;
  const int lane = threadIdx.x & 31;
  const int col = lane & 15;
  const int rowoff = (lane >> 4) << 3;
  const int b0 = bt * 16;

  v8f cst[4];
#pragma unroll
  for (int q = 0; q < 4; q++)
#pragma unroll
    for (int r = 0; r < 8; r++) cst[q][r] = 0.f;
  _Float16 hstash[4][8];

  for (int t = 0; t < S_LEN; ++t) {
    const int s = dir ? (S_LEN - 1 - t) : t;
    if (t + 1 < S_LEN) {
      int sn = dir ? (S_LEN - 2 - t) : (t + 1);
      __builtin_prefetch(xg + (size_t)(sn * BATCH + b0 + col) * 2048 + wave * 256, 0, 1);
    }
#pragma unroll
    for (int q = 0; q < 4; q++) {
      const int jt = wave * 4 + q;     // 0..31 over the 512-wide hidden
      v8f acc[4];
#pragma unroll
      for (int G = 0; G < 4; G++)
#pragma unroll
        for (int r = 0; r < 8; r++)
          acc[G][r] = xg[(size_t)(s * BATCH + b0 + r + rowoff) * 2048 +
                         G * H_W + jt * 16 + col];

#pragma unroll 4
      for (int k = 0; k < 16; k++) {
        v16h a = load_a_frag(hl, H_W, k * 32);
#pragma unroll
        for (int G = 0; G < 4; G++)
          acc[G] = wmma_f16(a, load_b_frag(pkWhh, (G * 32 + jt) * 16 + k), acc[G]);
      }

#pragma unroll
      for (int r = 0; r < 8; r++) {
        float ci = cst[q][r];
        ci = sigm(acc[1][r]) * ci + sigm(acc[0][r]) * tanh_(acc[2][r]);
        cst[q][r] = ci;
        float hv = sigm(acc[3][r]) * tanh_(ci);
        size_t o = (size_t)(s * BATCH + b0 + r + rowoff) * (2 * H_W) +
                   dir * H_W + jt * 16 + col;
        out_h[o] = hv;
        out_c[o] = ci;
        hstash[q][r] = (_Float16)hv;
      }
    }
    __syncthreads();  // all waves finished reading h across full K
#pragma unroll
    for (int q = 0; q < 4; q++)
#pragma unroll
      for (int r = 0; r < 8; r++)
        hl[(r + rowoff) * H_W + (wave * 4 + q) * 16 + col] = hstash[q][r];
    __syncthreads();
  }
}

// ---------------------------------------------------------------------------
extern "C" void kernel_launch(void* const* d_in, const int* in_sizes, int n_in,
                              void* d_out, int out_size, void* d_ws, size_t ws_size,
                              hipStream_t stream) {
  const int*   words = (const int*)d_in[0];
  const int*   chars = (const int*)d_in[1];
  const float* wemb  = (const float*)d_in[2];
  const float* cemb  = (const float*)d_in[3];
  const float* cWihF = (const float*)d_in[4];
  const float* cWhhF = (const float*)d_in[5];
  const float* cbF   = (const float*)d_in[6];
  const float* cWihB = (const float*)d_in[7];
  const float* cWhhB = (const float*)d_in[8];
  const float* cbB   = (const float*)d_in[9];
  const float* wWih  = (const float*)d_in[10];
  const float* wWhh  = (const float*)d_in[11];
  const float* wb    = (const float*)d_in[12];

  char* ws = (char*)d_ws;
  size_t off = 0;
  auto take = [&](size_t bytes) -> char* {
    char* p = ws + off;
    off = (off + bytes + 255) & ~(size_t)255;
    return p;
  };
  _Float16* pkCihF = (_Float16*)take((size_t)32768 * 2);
  _Float16* pkChhF = (_Float16*)take((size_t)65536 * 2);
  _Float16* pkCihB = (_Float16*)take((size_t)32768 * 2);
  _Float16* pkChhB = (_Float16*)take((size_t)65536 * 2);
  _Float16* pkWih  = (_Float16*)take((size_t)128 * 18 * 512 * 2);
  _Float16* pkWhh  = (_Float16*)take((size_t)128 * 16 * 512 * 2);
  _Float16* cemb16 = (_Float16*)take((size_t)128 * 64 * 2);
  _Float16* cfeat  = (_Float16*)take((size_t)8192 * 256 * 2);
  float*    xg     = (float*)take((size_t)8192 * 2048 * 4);

  // Weight prep (f16 fragment packing)
  pack_b_kernel<<<128, 256, 0, stream>>>(cWihF, pkCihF, 512, 64, 64);
  pack_b_kernel<<<128, 256, 0, stream>>>(cWhhF, pkChhF, 512, 128, 128);
  pack_b_kernel<<<128, 256, 0, stream>>>(cWihB, pkCihB, 512, 64, 64);
  pack_b_kernel<<<128, 256, 0, stream>>>(cWhhB, pkChhB, 512, 128, 128);
  pack_b_kernel<<<512, 256, 0, stream>>>(wWih, pkWih, 2048, 556, KPAD_WIH);
  pack_b_kernel<<<512, 256, 0, stream>>>(wWhh, pkWhh, 2048, 512, 512);
  conv_f16_kernel<<<32, 256, 0, stream>>>(cemb, cemb16, 128 * 64);

  // Char bi-LSTM: 198 KB dynamic LDS (within the 320 KB WGP LDS)
  const int CHAR_SMEM = (32768 + 65536 + 16 * H_C + 16 * E_C) * 2;
  (void)hipFuncSetAttribute((const void*)char_lstm_kernel,
                            hipFuncAttributeMaxDynamicSharedMemorySize, CHAR_SMEM);
  char_lstm_kernel<<<dim3(512, 2), 256, CHAR_SMEM, stream>>>(
      chars, cemb16, pkCihF, pkChhF, pkCihB, pkChhB, cbF, cbB, cfeat);

  // Word input projection (shared by both directions)
  word_xg_kernel<<<dim3(512, 16), 256, 0, stream>>>(words, wemb, cfeat, pkWih,
                                                    wb, xg);

  // Word bi-directional recurrence, writing hidden/cell directly
  float* out_h = (float*)d_out;
  float* out_c = out_h + (size_t)S_LEN * BATCH * 2 * H_W;
  word_lstm_kernel<<<dim3(8, 2), 256, 0, stream>>>(xg, pkWhh, out_h, out_c);
}
